// PointerNetForTriangles_30580167147637
// MI455X (gfx1250) — matrix operand order repaired and checked by reading, not compiled, not run on hardware
//
#include <hip/hip_runtime.h>
#include <hip/hip_bf16.h>

// ---------------------------------------------------------------------------
// PointerNet (encoder LSTM 1024 steps + pointer decoder 40 steps) for gfx1250
// B=64, SEQ=1024, D_IN=2, H=256, MAX_STEPS=40
// Strategy: persistent kernels with device-scope barriers; all recurrent
// matmuls via v_wmma_f32_16x16x32_bf16; encoder weights register-resident;
// all fragment/bulk traffic as b128; attention sweep as coalesced b64.
// ---------------------------------------------------------------------------

typedef __attribute__((ext_vector_type(16))) __bf16 v16bf;
typedef __attribute__((ext_vector_type(8)))  float  v8f;

__device__ inline unsigned short f2bf(float f) {
    unsigned x = __float_as_uint(f);
    unsigned r = x + 0x7fffu + ((x >> 16) & 1u);   // round-to-nearest-even
    return (unsigned short)(r >> 16);
}
__device__ inline float bf2f(unsigned short u) {
    return __uint_as_float(((unsigned)u) << 16);
}
__device__ inline float sigf(float x) { return 1.f / (1.f + __expf(-x)); }

__device__ inline v8f wmma_bf16(v16bf a, v16bf b, v8f c) {
    return __builtin_amdgcn_wmma_f32_16x16x32_bf16(
        /*neg_a=*/false, a, /*neg_b=*/false, b,
        /*c_mod=*/(short)0, c, /*reuse_a=*/false, /*reuse_b=*/false);
}

// A fragment (16x32, MxK) from LDS row-major bf16, row stride S2 (in u32).
// Lane layout: row M = lane%16; dword v holds K pair at
//   off(u32) = ((v&4)?8:0) + (lane/16)*4 + (v&3)  -> two contiguous 16B runs.
__device__ inline v16bf load_A_lds(const unsigned* hU, int S2, int mbase, int k0, int lane) {
    int m = mbase + (lane & 15);
    int half = (lane >> 4) & 1;
    const uint4* p = (const uint4*)(hU + m * S2 + (k0 >> 1) + half * 4);
    union { v16bf v; uint4 q[2]; } f;
    f.q[0] = p[0];          // dwords 0..3  (ds_load_b128)
    f.q[1] = p[2];          // dwords 8..11 relative (= +8 u32)
    return f.v;
}

// B fragment (32x16, KxN): B[k][n] = W[nbase+n][k0+k], W row-major bf16
// (row stride K2 in u32). Lane n = lane%16; dwords at off(u32) = ((v&4)?8:0) + (lane/16)*4 + (v&3).
__device__ inline v16bf load_B_glob(const unsigned* wU, int K2, int nbase, int k0, int lane) {
    int n = nbase + (lane & 15);
    int half = (lane >> 4) & 1;
    const uint4* p = (const uint4*)(wU + n * K2 + (k0 >> 1) + half * 4);
    union { v16bf v; uint4 q[2]; } f;
    f.q[0] = p[0];          // global_load_b128
    f.q[1] = p[2];          // global_load_b128
    return f.v;
}

// C (16x16 f32) store to LDS with row stride SC: VGPR r -> M = r + 8*(lane/16)
__device__ inline void store_C_lds(float* g, int SC, int mbase, int nbase, int lane, v8f c) {
    int n = nbase + (lane & 15);
    int mo = (lane >> 4) * 8;
#pragma unroll
    for (int r = 0; r < 8; ++r) g[(mbase + mo + r) * SC + n] = c[r];
}

// Device-scope sense barrier over nwg workgroups; cnt/gen zeroed by prep.
__device__ inline void grid_barrier(unsigned* cnt, unsigned* gen, unsigned nwg, unsigned phase) {
    __threadfence();
    __syncthreads();
    if (threadIdx.x == 0) {
        unsigned prev = __hip_atomic_fetch_add(cnt, 1u, __ATOMIC_ACQ_REL, __HIP_MEMORY_SCOPE_AGENT);
        if (prev == nwg * (phase + 1u) - 1u) {
            __hip_atomic_store(gen, phase + 1u, __ATOMIC_RELEASE, __HIP_MEMORY_SCOPE_AGENT);
        } else {
            while (__hip_atomic_load(gen, __ATOMIC_ACQUIRE, __HIP_MEMORY_SCOPE_AGENT) < phase + 1u)
                __builtin_amdgcn_s_sleep(2);
        }
    }
    __syncthreads();
    __threadfence();
}

// ---------------------------------------------------------------------------
// Prep: bf16 weight conversion, combined decoder weight [W_ih|W_hh] (K=1024),
// fused biases, start-token / end-node broadcast, barrier reset.
// ---------------------------------------------------------------------------
__global__ void __launch_bounds__(256) prep_kernel(
    const float* __restrict__ encWhh, const float* __restrict__ encbih,
    const float* __restrict__ encbhh, const float* __restrict__ decWih,
    const float* __restrict__ decWhh, const float* __restrict__ decbih,
    const float* __restrict__ decbhh, const float* __restrict__ Wq,
    const float* __restrict__ endNode, const float* __restrict__ startTok,
    unsigned short* __restrict__ encWhhB, unsigned short* __restrict__ WcombB,
    unsigned short* __restrict__ WqB, float* __restrict__ enc_bias,
    float* __restrict__ dec_bias, unsigned short* __restrict__ inpfull,
    unsigned short* __restrict__ encT, unsigned* __restrict__ bars)
{
    long i = (long)blockIdx.x * 256 + threadIdx.x;
    const long N0 = 262144, N1 = 1048576, N2 = 65536, N3 = 1024, N4 = 1024,
               N5 = 49152, N6 = 16384, N7 = 64;
    if (i < N0) { encWhhB[i] = f2bf(encWhh[i]); return; } i -= N0;
    if (i < N1) {
        long n = i >> 10, k = i & 1023;
        float v = (k < 768) ? decWih[n * 768 + k] : decWhh[n * 256 + (k - 768)];
        WcombB[i] = f2bf(v); return;
    } i -= N1;
    if (i < N2) { WqB[i] = f2bf(Wq[i]); return; } i -= N2;
    if (i < N3) { enc_bias[i] = encbih[i] + encbhh[i]; return; } i -= N3;
    if (i < N4) { dec_bias[i] = decbih[i] + decbhh[i]; return; } i -= N4;
    if (i < N5) { long b = i / 768, k = i % 768; inpfull[b * 1024 + k] = f2bf(startTok[k]); return; } i -= N5;
    if (i < N6) { long b = i >> 8, k = i & 255; encT[(b * 256 + k) * 1040 + 1024] = f2bf(endNode[k]); return; } i -= N6;
    if (i < N7) { bars[i] = 0u; return; }
}

// ---------------------------------------------------------------------------
// Encoder: 16 persistent WGs x 256 threads. WG jt owns hidden columns
// jt*16..+15 (and gate columns g*256+j for all four gates). W_hh bf16 slice
// is register-resident (8 B-fragments per wave). h round-trips through L2
// once per step (b128 both ways); grid barrier per step.
// ---------------------------------------------------------------------------
__global__ void __launch_bounds__(256) enc_kernel(
    const float* __restrict__ x,            // [64][1024][2]
    const float* __restrict__ Wih,          // [1024][2]
    const unsigned* __restrict__ WhhU,      // bf16 [1024][256] as u32 pairs
    const float* __restrict__ bias,         // [1024] fused
    unsigned short* __restrict__ h_ws,      // [64][256] bf16
    float* __restrict__ c_ws,               // [64][256]
    unsigned short* __restrict__ inpfull,   // [64][1024] bf16 (decoder input)
    unsigned short* __restrict__ encT,      // [64][256][1040] bf16 (enc_out^T)
    unsigned* barCnt, unsigned* barGen)
{
    extern __shared__ char smem[];
    unsigned* hU  = (unsigned*)smem;                                  // 8192 u32
    float* gatesS = (float*)(smem + 32768);                           // 4096 f32
    float* cS     = (float*)(smem + 32768 + 16384);                   // 1024 f32
    float* wihS   = (float*)(smem + 32768 + 16384 + 4096);            // 2048 f32
    float* biasS  = (float*)(smem + 32768 + 16384 + 4096 + 8192);     // 1024 f32

    const int tid = threadIdx.x, lane = tid & 31, wave = tid >> 5;
    const int jt = blockIdx.x;           // 0..15
    const int nt = wave >> 1;            // gate 0..3
    const int mb = (wave & 1) * 32;      // batch-row base (tiles mb, mb+16)

    for (int i = tid; i < 8192; i += 256) hU[i] = 0u;       // h0 = 0
    for (int i = tid; i < 1024; i += 256) cS[i] = 0.f;      // c0 = 0
    for (int i = tid; i < 2048; i += 256) wihS[i] = Wih[i];
    for (int i = tid; i < 1024; i += 256) biasS[i] = bias[i];

    const int nbase = nt * 256 + jt * 16;
    v16bf Bfr[8];                        // W_hh slice: register-resident
#pragma unroll
    for (int kt = 0; kt < 8; ++kt) Bfr[kt] = load_B_glob(WhhU, 128, nbase, kt * 32, lane);
    __syncthreads();

    unsigned phase = 0;
    for (int t = 0; t < 1024; ++t) {
        v8f C0 = {}; v8f C1 = {};
#pragma unroll
        for (int kt = 0; kt < 8; ++kt) {
            v16bf A0 = load_A_lds(hU, 128, mb,      kt * 32, lane);
            v16bf A1 = load_A_lds(hU, 128, mb + 16, kt * 32, lane);
            C0 = wmma_bf16(A0, Bfr[kt], C0);
            C1 = wmma_bf16(A1, Bfr[kt], C1);
        }
        store_C_lds(gatesS, 64, mb,      nt * 16, lane, C0);
        store_C_lds(gatesS, 64, mb + 16, nt * 16, lane, C1);
        __syncthreads();

        { // pointwise LSTM cell on our 16-column slice (thread -> 4 elems)
            int base = tid * 4;
            int b = base >> 4, jl0 = base & 15;
            float x0 = x[(b * 1024 + t) * 2 + 0];
            float x1 = x[(b * 1024 + t) * 2 + 1];
#pragma unroll
            for (int q = 0; q < 4; ++q) {
                int jl = jl0 + q, jg = jt * 16 + jl;
                float g4[4];
#pragma unroll
                for (int g = 0; g < 4; ++g) {
                    int n = g * 256 + jg;
                    g4[g] = gatesS[b * 64 + g * 16 + jl]
                          + x0 * wihS[n * 2 + 0] + x1 * wihS[n * 2 + 1] + biasS[n];
                }
                float ci = cS[b * 16 + jl];
                float c2 = sigf(g4[1]) * ci + sigf(g4[0]) * tanhf(g4[2]);
                float h2 = sigf(g4[3]) * tanhf(c2);
                cS[b * 16 + jl] = c2;
                unsigned short hb = f2bf(h2);
                h_ws[b * 256 + jg] = hb;
                encT[(size_t)(b * 256 + jg) * 1040 + t] = hb;
                if (t == 1023) {                // hand state to decoder
                    c_ws[b * 256 + jg] = c2;
                    inpfull[b * 1024 + 768 + jg] = hb;
                }
            }
        }
        grid_barrier(barCnt, barGen, 16, phase); ++phase;
        // refresh full h into LDS as b128
        {
            const uint4* src = (const uint4*)h_ws;
            uint4* dst = (uint4*)hU;
            for (int i = tid; i < 2048; i += 256) dst[i] = src[i];
        }
        __syncthreads();
    }
}

// ---------------------------------------------------------------------------
// Decoder: 64 persistent WGs x 256 threads, 40 steps, 3 grid barriers/step.
//   A (WG 0..15): gates = [inp|h](64x1024) @ Wcomb^T, WMMA K=1024 -> h,c
//   B (WG 0..15): q = h @ Wq^T + bq, WMMA K=256
//   C/D/E (WG b): logits[b] = enc^T q  -> top-3 -> gather next input
// ---------------------------------------------------------------------------
__global__ void __launch_bounds__(256) dec_kernel(
    const unsigned* __restrict__ WcombU,     // bf16 [1024][1024] as u32 pairs
    const float* __restrict__ dec_bias,      // [1024]
    const unsigned* __restrict__ WqU,        // bf16 [256][256] as u32 pairs
    const float* __restrict__ bq,            // [256]
    unsigned short* __restrict__ inpfull,    // [64][1024] bf16 = inp(768)|h(256)
    float* __restrict__ c_ws,                // [64][256]
    unsigned short* __restrict__ q_ws,       // [64][256] bf16
    const unsigned short* __restrict__ encT, // [64][256][1040] bf16
    float* __restrict__ out_logits,          // [40][64][1025]
    int* __restrict__ out_idx,               // [40][64][3]
    unsigned* barCnt, unsigned* barGen)
{
    extern __shared__ char smem[];
    unsigned* inpU = (unsigned*)smem;                 // 32768 u32 (128KB)
    float* gatesS  = (float*)(smem + 131072);         // 4096 f32
    float* qv      = (float*)(smem + 147456);         // 256 f32
    float* logitsS = (float*)(smem + 148480);         // 1056 f32
    float* redv    = (float*)(smem + 152704);         // 256 f32
    int*   redi    = (int*)  (smem + 153728);         // 256 i32
    int*   topsS   = (int*)  (smem + 154752);         // 4 i32

    const int tid = threadIdx.x, lane = tid & 31, wave = tid >> 5;
    const int wg = blockIdx.x;   // 0..63
    const int b  = wg;           // attention row for phases C/D/E
    unsigned phase = 0;

    for (int s = 0; s < 40; ++s) {
        // ---- Phase A: LSTM gates + cell (WGs 0..15) ----
        if (wg < 16) {
            {   // load [inp|h] (128KB) into LDS as b128
                const uint4* src = (const uint4*)inpfull;
                uint4* dst = (uint4*)inpU;
                for (int i = tid; i < 8192; i += 256) dst[i] = src[i];
            }
            __syncthreads();
            const int jt = wg, nt = wave >> 1, mb = (wave & 1) * 32;
            const int nbase = nt * 256 + jt * 16;
            v8f C0 = {}, C1 = {};
            for (int kt = 0; kt < 32; ++kt) {
                v16bf Bf = load_B_glob(WcombU, 512, nbase, kt * 32, lane);
                v16bf A0 = load_A_lds(inpU, 512, mb,      kt * 32, lane);
                v16bf A1 = load_A_lds(inpU, 512, mb + 16, kt * 32, lane);
                C0 = wmma_bf16(A0, Bf, C0);
                C1 = wmma_bf16(A1, Bf, C1);
            }
            store_C_lds(gatesS, 64, mb,      nt * 16, lane, C0);
            store_C_lds(gatesS, 64, mb + 16, nt * 16, lane, C1);
            __syncthreads();
            int base = tid * 4, bb = base >> 4, jl0 = base & 15;
#pragma unroll
            for (int q = 0; q < 4; ++q) {
                int jl = jl0 + q, jg = jt * 16 + jl;
                float gi = gatesS[bb * 64 +  0 + jl] + dec_bias[      jg];
                float gf = gatesS[bb * 64 + 16 + jl] + dec_bias[256 + jg];
                float gg = gatesS[bb * 64 + 32 + jl] + dec_bias[512 + jg];
                float go = gatesS[bb * 64 + 48 + jl] + dec_bias[768 + jg];
                float ci = c_ws[bb * 256 + jg];
                float c2 = sigf(gf) * ci + sigf(gi) * tanhf(gg);
                float h2 = sigf(go) * tanhf(c2);
                c_ws[bb * 256 + jg] = c2;
                inpfull[bb * 1024 + 768 + jg] = f2bf(h2);
            }
        }
        grid_barrier(barCnt, barGen, 64, phase); ++phase;

        // ---- Phase B: q = h @ Wq^T + bq (WGs 0..15, 16 cols each) ----
        if (wg < 16) {
            {   // refresh h region (cols 768..895) as b128
                const uint4* src = (const uint4*)inpfull;
                uint4* dst = (uint4*)inpU;
                for (int i = tid; i < 1024; i += 256) {
                    int row = i >> 4, col = i & 15;
                    dst[row * 128 + 96 + col] = src[row * 128 + 96 + col];
                }
            }
            __syncthreads();
            if (wave < 4) {
                const int mbq = wave * 16;
                v8f C = {};
#pragma unroll
                for (int kt = 0; kt < 8; ++kt) {
                    v16bf A  = load_A_lds(inpU, 512, mbq, 768 + kt * 32, lane);
                    v16bf Bf = load_B_glob(WqU, 128, wg * 16, kt * 32, lane);
                    C = wmma_bf16(A, Bf, C);
                }
                store_C_lds(gatesS, 16, mbq, 0, lane, C);
            }
            __syncthreads();
            for (int e = tid; e < 1024; e += 256) {
                int bb = e >> 4, jl = e & 15, col = wg * 16 + jl;
                q_ws[bb * 256 + col] = f2bf(gatesS[bb * 16 + jl] + bq[col]);
            }
        }
        grid_barrier(barCnt, barGen, 64, phase); ++phase;

        // ---- Phase C: logits[b][n] = sum_k encT[b][k][n] * q[b][k] ----
        qv[tid] = bf2f(q_ws[b * 256 + tid]);
        __syncthreads();
        {   // thread owns columns n = 4*tid .. 4*tid+3; one b64 load per k
            const uint2* erow = (const uint2*)(encT + (size_t)b * 256 * 1040);
            float a0 = 0.f, a1 = 0.f, a2 = 0.f, a3 = 0.f;
            for (int k = 0; k < 256; ++k) {
                uint2 w = erow[(size_t)k * 260 + tid];
                float q = qv[k];
                a0 += __uint_as_float(w.x << 16)          * q;
                a1 += __uint_as_float(w.x & 0xffff0000u)  * q;
                a2 += __uint_as_float(w.y << 16)          * q;
                a3 += __uint_as_float(w.y & 0xffff0000u)  * q;
            }
            int n0 = tid * 4;
            logitsS[n0] = a0; logitsS[n0 + 1] = a1;
            logitsS[n0 + 2] = a2; logitsS[n0 + 3] = a3;
            float* outp = out_logits + ((size_t)s * 64 + b) * 1025 + n0;
            outp[0] = a0; outp[1] = a1; outp[2] = a2; outp[3] = a3;
        }
        // n = 1024 (end node) via 256-way reduction over k
        {
            const unsigned short* ep = encT + (size_t)b * 256 * 1040 + 1024;
            redv[tid] = bf2f(ep[(size_t)tid * 1040]) * qv[tid];
        }
        __syncthreads();
        for (int off = 128; off > 0; off >>= 1) {
            if (tid < off) redv[tid] += redv[tid + off];
            __syncthreads();
        }
        if (tid == 0) {
            logitsS[1024] = redv[0];
            out_logits[((size_t)s * 64 + b) * 1025 + 1024] = redv[0];
        }
        __syncthreads();

        // ---- Phase D: top-3 (ties -> lower index, like jax top_k) ----
        int e0 = -1, e1 = -1;
        for (int p = 0; p < 3; ++p) {
            float bv = -3.4e38f; int bi = 0x7fffffff;
            for (int n = tid; n < 1025; n += 256) {
                if (n == e0 || n == e1) continue;
                float v = logitsS[n];
                if (v > bv || (v == bv && n < bi)) { bv = v; bi = n; }
            }
            redv[tid] = bv; redi[tid] = bi;
            __syncthreads();
            for (int off = 128; off > 0; off >>= 1) {
                if (tid < off) {
                    float vo = redv[tid + off]; int io = redi[tid + off];
                    if (vo > redv[tid] || (vo == redv[tid] && io < redi[tid])) {
                        redv[tid] = vo; redi[tid] = io;
                    }
                }
                __syncthreads();
            }
            int win = redi[0];
            if (tid == 0) { topsS[p] = win; out_idx[((size_t)s * 64 + b) * 3 + p] = win; }
            __syncthreads();
            if (p == 0) e0 = win; else if (p == 1) e1 = win;
        }

        // ---- Phase E: gather top-3 vectors -> next input (bf16 copy) ----
        for (int e = tid; e < 768; e += 256) {
            int slot = e >> 8, k = e & 255;
            int id = topsS[slot];
            inpfull[(size_t)b * 1024 + slot * 256 + k] =
                encT[((size_t)b * 256 + k) * 1040 + id];
        }
        grid_barrier(barCnt, barGen, 64, phase); ++phase;
    }
}

// ---------------------------------------------------------------------------
// Host launcher
// ---------------------------------------------------------------------------
extern "C" void kernel_launch(void* const* d_in, const int* in_sizes, int n_in,
                              void* d_out, int out_size, void* d_ws, size_t ws_size,
                              hipStream_t stream) {
    const float* x        = (const float*)d_in[0];
    const float* encWih   = (const float*)d_in[1];
    const float* encWhh   = (const float*)d_in[2];
    const float* encbih   = (const float*)d_in[3];
    const float* encbhh   = (const float*)d_in[4];
    const float* decWih   = (const float*)d_in[5];
    const float* decWhh   = (const float*)d_in[6];
    const float* decbih   = (const float*)d_in[7];
    const float* decbhh   = (const float*)d_in[8];
    const float* Wq       = (const float*)d_in[9];
    const float* bq       = (const float*)d_in[10];
    const float* endNode  = (const float*)d_in[11];
    const float* startTok = (const float*)d_in[12];

    char* ws = (char*)d_ws;
    // workspace layout (bytes)
    unsigned short* encWhhB = (unsigned short*)(ws + 0);          // 524288
    unsigned short* WcombB  = (unsigned short*)(ws + 524288);     // 2097152
    unsigned short* WqB     = (unsigned short*)(ws + 2621440);    // 131072
    float*          encBias = (float*)(ws + 2752512);             // 4096
    float*          decBias = (float*)(ws + 2756608);             // 4096
    unsigned short* h_ws    = (unsigned short*)(ws + 2760704);    // 32768
    float*          c_ws    = (float*)(ws + 2793472);             // 65536
    unsigned short* inpfull = (unsigned short*)(ws + 2859008);    // 131072
    unsigned short* q_ws    = (unsigned short*)(ws + 2990080);    // 32768
    unsigned*       bars    = (unsigned*)(ws + 3022848);          // 256
    unsigned short* encT    = (unsigned short*)(ws + 3023104);    // 34078720
    const size_t need = 3023104 + 34078720;                       // ~37.1 MB
    if (ws_size < need) return;

    float* out_logits = (float*)d_out;                 // 40*64*1025 floats
    int*   out_idx    = (int*)((float*)d_out + (size_t)40 * 64 * 1025);

    // prep: weight conversion + broadcast + barrier reset
    {
        long total = 262144 + 1048576 + 65536 + 1024 + 1024 + 49152 + 16384 + 64;
        int blocks = (int)((total + 255) / 256);
        prep_kernel<<<blocks, 256, 0, stream>>>(
            encWhh, encbih, encbhh, decWih, decWhh, decbih, decbhh, Wq,
            endNode, startTok, encWhhB, WcombB, WqB, encBias, decBias,
            inpfull, encT, bars);
    }
    // encoder: 16 persistent WGs, 64 KB LDS each
    enc_kernel<<<16, 256, 65536, stream>>>(
        x, encWih, (const unsigned*)encWhhB, encBias,
        h_ws, c_ws, inpfull, encT, &bars[0], &bars[1]);
    // decoder: 64 persistent WGs, ~151 KB LDS each
    dec_kernel<<<64, 256, 154768, stream>>>(
        (const unsigned*)WcombB, decBias, (const unsigned*)WqB, bq,
        inpfull, c_ws, q_ws, encT, out_logits, out_idx, &bars[2], &bars[3]);
}
